// Generator_17454747091716
// MI455X (gfx1250) — compile-verified, matrix operand order checked by reading, not run on hardware
//
#include <hip/hip_runtime.h>
#include <hip/hip_bf16.h>

// ---------------------------------------------------------------------------
// CDNA5 (gfx1250) transformer-encoder forward pass.
// All GEMMs run through V_WMMA_F32_16X16X32_BF16 (f32 accumulate).
// Operands are pre-converted to bf16 once (halves HBM traffic); all B
// operands are kept in B^T [N][K] form so both tiles stage with contiguous
// 16B copies -> GLOBAL_LOAD_ASYNC_TO_LDS_B128 when the toolchain exposes it.
// ---------------------------------------------------------------------------

typedef __attribute__((ext_vector_type(16))) __bf16 v16bf;
typedef __attribute__((ext_vector_type(8)))  __bf16 v8bf;
typedef __attribute__((ext_vector_type(4)))  __bf16 v4bf;
typedef __attribute__((ext_vector_type(8)))  float  v8f;

// The async-to-LDS builtin takes int4* in AS1 (global) / AS3 (LDS); clang
// prints the pointee as __attribute__((vector_size(16))) int, so match that.
typedef int v4i_vs __attribute__((vector_size(16)));
typedef __attribute__((address_space(1))) v4i_vs as1_v4i;
typedef __attribute__((address_space(3))) v4i_vs as3_v4i;

#if defined(__has_builtin)
#if __has_builtin(__builtin_amdgcn_global_load_async_to_lds_b128)
#define USE_ASYNC_LDS 1
#endif
#endif
#ifndef USE_ASYNC_LDS
#define USE_ASYNC_LDS 0
#endif

#define D_MODEL  384
#define N_TOK    1024
#define N_HEADS  6
#define HEAD     64
#define N_LAYERS 6
#define BATCH    4
#define VOCAB    8192
#define ROWS     (BATCH * N_TOK)      // 4096

// GEMM tiling: 64x64 block tile, BK=32, 4 waves (128 threads), wave tile 32x32.
#define BM   64
#define BN   64
#define BK   32
#define LDSS 40   // padded LDS row stride (bf16); 80B rows keep 16B alignment

// Epilogues
#define EPI_NONE      0
#define EPI_BIAS      1
#define EPI_BIAS_GELU 2
#define EPI_BIAS_RES  3
#define EPI_POSENC    4
// Output kinds
#define OUT_F32   0
#define OUT_BF16  1
#define OUT_BF16T 2   // bf16, transposed store (C[n][m])

// ---------------------------------------------------------------------------
// 16B tile-staging copy: async-to-LDS on CDNA5 toolchains, sync fallback.
// ---------------------------------------------------------------------------
__device__ __forceinline__ void stage_b128(const __bf16* g, __bf16* l) {
#if USE_ASYNC_LDS
  __builtin_amdgcn_global_load_async_to_lds_b128((as1_v4i*)g, (as3_v4i*)l, 0, 0);
#else
  *(v8bf*)l = *(const v8bf*)g;
#endif
}

__device__ __forceinline__ void stage_fence() {
#if USE_ASYNC_LDS
#if __has_builtin(__builtin_amdgcn_s_wait_asynccnt)
  __builtin_amdgcn_s_wait_asynccnt(0);
#else
  asm volatile("s_wait_asynccnt 0x0" ::: "memory");
#endif
#endif
}

// ---------------------------------------------------------------------------
// Fragment loader: ISA 16-bit A-matrix 16x32 layout (CDNA5 ISA 7.12.2).
// Lanes 0-15: K=[0..7]+[16..23]; lanes 16-31: K=[8..15]+[24..31].
// Each half = 8 consecutive bf16 = one ds_load_b128.
// ---------------------------------------------------------------------------
__device__ __forceinline__ v16bf load_frag(const __bf16* __restrict__ sm,
                                           int rowBase, int lane) {
  const int r  = rowBase + (lane & 15);
  const int kh = (lane & 16) >> 1;
  const __bf16* p = sm + r * LDSS + kh;
  v8bf lo = *(const v8bf*)(p);
  v8bf hi = *(const v8bf*)(p + 16);
  v16bf out;
#pragma unroll
  for (int i = 0; i < 8; ++i) { out[i] = lo[i]; out[i + 8] = hi[i]; }
  return out;
}

__device__ __forceinline__ float gelu_exact(float x) {
  return 0.5f * x * (1.0f + erff(x * 0.70710678118654752f));
}

// ---------------------------------------------------------------------------
// Generic bf16 WMMA GEMM:  C[z] = alpha * A[z] * Bt[z]^T  (+ epilogue)
//   A : [M][K] bf16, lda     Bt : [N][K] bf16, ldb
// z = b*H + h with independent b/h pointer strides.  M%64==N%64==0, K%32==0
// for every GEMM in this model, so EXEC is all-ones at each WMMA.
// ---------------------------------------------------------------------------
template <int EPI, int OUTK>
__global__ __launch_bounds__(128) void gemm_bf16_wmma_kernel(
    const __bf16* __restrict__ A, const __bf16* __restrict__ Bt,
    const float* __restrict__ bias, const float* __restrict__ res,
    void* __restrict__ Cv,
    int N, int K, int lda, int ldb, int ldc,
    int H, long long sAb, long long sAh, long long sBb, long long sBh,
    long long sCb, long long sCh, long long sBiasH, float alpha, int nTok) {
  const int zb = blockIdx.z / H;
  const int zh = blockIdx.z % H;
  A  += (long long)zb * sAb + (long long)zh * sAh;
  Bt += (long long)zb * sBb + (long long)zh * sBh;
  const long long coff = (long long)zb * sCb + (long long)zh * sCh;
  if (EPI != EPI_NONE) bias += (long long)zh * sBiasH;

  __shared__ __align__(16) __bf16 smA[BM * LDSS];
  __shared__ __align__(16) __bf16 smB[BN * LDSS];

  const int tid   = threadIdx.x;
  const int lane  = tid & 31;
  const int wid   = tid >> 5;
  const int wr    = (wid >> 1) * 32;
  const int wc    = (wid & 1) * 32;
  const int tileM = blockIdx.y * BM;
  const int tileN = blockIdx.x * BN;

  v8f zero = {};
  v8f acc[2][2];
#pragma unroll
  for (int i = 0; i < 2; ++i)
#pragma unroll
    for (int j = 0; j < 2; ++j) acc[i][j] = zero;

  for (int k0 = 0; k0 < K; k0 += BK) {
    __syncthreads();
    // stage A and Bt tiles: 64 rows x 32 bf16 = 256 chunks of 16B each
#pragma unroll
    for (int t = 0; t < 2; ++t) {
      const int i = tid + t * 128;
      const int r = i >> 2, c = (i & 3) << 3;
      stage_b128(&A[(long long)(tileM + r) * lda + k0 + c], &smA[r * LDSS + c]);
    }
#pragma unroll
    for (int t = 0; t < 2; ++t) {
      const int i = tid + t * 128;
      const int r = i >> 2, c = (i & 3) << 3;
      stage_b128(&Bt[(long long)(tileN + r) * ldb + k0 + c], &smB[r * LDSS + c]);
    }
    if (k0 + BK < K) {  // pull next K tile toward L2 (global_prefetch_b8)
      __builtin_prefetch(&A[(long long)(tileM + (tid >> 1)) * lda + k0 + BK], 0, 1);
      __builtin_prefetch(&Bt[(long long)(tileN + (tid >> 1)) * ldb + k0 + BK], 0, 1);
    }
    stage_fence();
    __syncthreads();

    v16bf af[2], bf[2];
    af[0] = load_frag(smA, wr,      lane);
    af[1] = load_frag(smA, wr + 16, lane);
    bf[0] = load_frag(smB, wc,      lane);
    bf[1] = load_frag(smB, wc + 16, lane);
#pragma unroll
    for (int fr = 0; fr < 2; ++fr)
#pragma unroll
      for (int fc = 0; fc < 2; ++fc)
        acc[fr][fc] = __builtin_amdgcn_wmma_f32_16x16x32_bf16(
            false, af[fr], false, bf[fc], (short)0, acc[fr][fc], false, false);
  }

  // epilogue: D layout = lanes 0-15 rows r, lanes 16-31 rows r+8, col = lane&15
  float* Cf = (float*)Cv;
  __bf16* Cb = (__bf16*)Cv;
  const int mHalf = (lane >> 4) * 8;
  const int nCol  = lane & 15;
#pragma unroll
  for (int fr = 0; fr < 2; ++fr)
#pragma unroll
    for (int fc = 0; fc < 2; ++fc)
#pragma unroll
      for (int r = 0; r < 8; ++r) {
        const int gm = tileM + wr + fr * 16 + mHalf + r;
        const int gn = tileN + wc + fc * 16 + nCol;
        float v = alpha * acc[fr][fc][r];
        if (EPI == EPI_BIAS) {
          v += bias[gn];
        } else if (EPI == EPI_BIAS_GELU) {
          v = gelu_exact(v + bias[gn]);
        } else if (EPI == EPI_BIAS_RES) {
          v += bias[gn] + res[(long long)gm * ldc + gn];
        } else if (EPI == EPI_POSENC) {
          const int pos = gm % nTok;
          const int de  = gn & ~1;
          float dt  = __expf(-(float)de * (9.2103403719761836f / 384.0f));
          float ang = (float)pos * dt;
          v += bias[gn] + ((gn & 1) ? __cosf(ang) : __sinf(ang));
        }
        if (OUTK == OUT_F32)        Cf[coff + (long long)gm * ldc + gn] = v;
        else if (OUTK == OUT_BF16)  Cb[coff + (long long)gm * ldc + gn] = (__bf16)v;
        else                        Cb[coff + (long long)gn * ldc + gm] = (__bf16)v;
      }
}

// ---------------------------------------------------------------------------
// fp32 -> bf16 bulk convert (float4 granular)
// ---------------------------------------------------------------------------
__global__ __launch_bounds__(256) void convert_bf16_kernel(
    const float* __restrict__ in, __bf16* __restrict__ out, long long n4) {
  const long long i = (long long)blockIdx.x * 256 + threadIdx.x;
  if (i < n4) {
    float4 f = ((const float4*)in)[i];
    v4bf o;
    o[0] = (__bf16)f.x; o[1] = (__bf16)f.y; o[2] = (__bf16)f.z; o[3] = (__bf16)f.w;
    ((v4bf*)out)[i] = o;
  }
}

// ---------------------------------------------------------------------------
// fp32 [K][N] -> bf16 [N][K] transpose (32x32 LDS tile, coalesced both sides)
// ---------------------------------------------------------------------------
__global__ __launch_bounds__(256) void transpose_bf16_kernel(
    const float* __restrict__ in, __bf16* __restrict__ out,
    int K, int N, long long sInZ, long long sOutZ) {
  in  += (long long)blockIdx.z * sInZ;
  out += (long long)blockIdx.z * sOutZ;
  __shared__ float t[32][33];
  const int n0 = blockIdx.x * 32, k0 = blockIdx.y * 32;
  const int tx = threadIdx.x & 31, ty = threadIdx.x >> 5;
#pragma unroll
  for (int i = 0; i < 4; ++i)
    t[ty + i * 8][tx] = in[(long long)(k0 + ty + i * 8) * N + n0 + tx];
  __syncthreads();
#pragma unroll
  for (int i = 0; i < 4; ++i)
    out[(long long)(n0 + ty + i * 8) * K + k0 + tx] = (__bf16)t[tx][ty + i * 8];
}

// ---------------------------------------------------------------------------
// LayerNorm over rows of length D, bf16 output (one 128-thread block per row)
// ---------------------------------------------------------------------------
__global__ __launch_bounds__(128) void layernorm_kernel(
    const float* __restrict__ x, const float* __restrict__ g,
    const float* __restrict__ b, __bf16* __restrict__ y, int D) {
  const int row = blockIdx.x;
  const float* xr = x + (long long)row * D;
  __bf16* yr = y + (long long)row * D;
  __shared__ float red[128];
  const int tid = threadIdx.x;

  float s = 0.f;
  for (int i = tid; i < D; i += 128) s += xr[i];
  red[tid] = s; __syncthreads();
  for (int off = 64; off > 0; off >>= 1) {
    if (tid < off) red[tid] += red[tid + off];
    __syncthreads();
  }
  const float mu = red[0] / (float)D;
  __syncthreads();

  float s2 = 0.f;
  for (int i = tid; i < D; i += 128) { float d = xr[i] - mu; s2 += d * d; }
  red[tid] = s2; __syncthreads();
  for (int off = 64; off > 0; off >>= 1) {
    if (tid < off) red[tid] += red[tid + off];
    __syncthreads();
  }
  const float rstd = rsqrtf(red[0] / (float)D + 1e-5f);

  for (int i = tid; i < D; i += 128)
    yr[i] = (__bf16)((xr[i] - mu) * rstd * g[i] + b[i]);
}

// ---------------------------------------------------------------------------
// Row softmax: fp32 scores in, bf16 probabilities out (256 threads per row)
// ---------------------------------------------------------------------------
__global__ __launch_bounds__(256) void softmax_kernel(
    float* __restrict__ s, __bf16* __restrict__ p, int L) {
  float* r = s + (long long)blockIdx.x * L;
  __bf16* pr = p + (long long)blockIdx.x * L;
  __shared__ float red[256];
  const int tid = threadIdx.x;

  float m = -3.402823466e38f;
  for (int i = tid; i < L; i += 256) m = fmaxf(m, r[i]);
  red[tid] = m; __syncthreads();
  for (int off = 128; off > 0; off >>= 1) {
    if (tid < off) red[tid] = fmaxf(red[tid], red[tid + off]);
    __syncthreads();
  }
  m = red[0]; __syncthreads();

  float sum = 0.f;
  for (int i = tid; i < L; i += 256) {
    float e = __expf(r[i] - m);
    r[i] = e;
    sum += e;
  }
  red[tid] = sum; __syncthreads();
  for (int off = 128; off > 0; off >>= 1) {
    if (tid < off) red[tid] += red[tid + off];
    __syncthreads();
  }
  const float inv = 1.0f / red[0];
  for (int i = tid; i < L; i += 256) pr[i] = (__bf16)(r[i] * inv);
}

// ---------------------------------------------------------------------------
// Host-side dispatch
// ---------------------------------------------------------------------------
static void launch_gemm(hipStream_t stream,
                        const __bf16* A, const __bf16* Bt, const float* bias,
                        const float* res, void* C,
                        int M, int N, int K, int lda, int ldb, int ldc,
                        int Z, int H,
                        long long sAb, long long sAh,
                        long long sBb, long long sBh,
                        long long sCb, long long sCh, long long sBiasH,
                        float alpha, int epi, int outk) {
  dim3 grid(N / BN, M / BM, Z);
  dim3 blk(128);
#define GO(EP, OK)                                                            \
  gemm_bf16_wmma_kernel<EP, OK><<<grid, blk, 0, stream>>>(                    \
      A, Bt, bias, res, C, N, K, lda, ldb, ldc, H, sAb, sAh, sBb, sBh,        \
      sCb, sCh, sBiasH, alpha, N_TOK)
  const int sel = epi * 3 + outk;
  switch (sel) {
    case EPI_NONE * 3 + OUT_F32:       GO(EPI_NONE, OUT_F32);       break;
    case EPI_NONE * 3 + OUT_BF16:      GO(EPI_NONE, OUT_BF16);      break;
    case EPI_BIAS * 3 + OUT_F32:       GO(EPI_BIAS, OUT_F32);       break;
    case EPI_BIAS * 3 + OUT_BF16:      GO(EPI_BIAS, OUT_BF16);      break;
    case EPI_BIAS * 3 + OUT_BF16T:     GO(EPI_BIAS, OUT_BF16T);     break;
    case EPI_BIAS_GELU * 3 + OUT_BF16: GO(EPI_BIAS_GELU, OUT_BF16); break;
    case EPI_BIAS_RES * 3 + OUT_F32:   GO(EPI_BIAS_RES, OUT_F32);   break;
    case EPI_POSENC * 3 + OUT_F32:     GO(EPI_POSENC, OUT_F32);     break;
  }
#undef GO
}

extern "C" void kernel_launch(void* const* d_in, const int* in_sizes, int n_in,
                              void* d_out, int out_size, void* d_ws, size_t ws_size,
                              hipStream_t stream) {
  const float* x     = (const float*)d_in[0];
  const float* Wemb  = (const float*)d_in[1];
  const float* bemb  = (const float*)d_in[2];
  const float* ln1_g = (const float*)d_in[3];
  const float* ln1_b = (const float*)d_in[4];
  const float* Wq    = (const float*)d_in[5];
  const float* bq    = (const float*)d_in[6];
  const float* Wk    = (const float*)d_in[7];
  const float* bk    = (const float*)d_in[8];
  const float* Wv    = (const float*)d_in[9];
  const float* bv    = (const float*)d_in[10];
  const float* Wo    = (const float*)d_in[11];
  const float* bo    = (const float*)d_in[12];
  const float* ln2_g = (const float*)d_in[13];
  const float* ln2_b = (const float*)d_in[14];
  const float* W1    = (const float*)d_in[15];
  const float* b1    = (const float*)d_in[16];
  const float* W2    = (const float*)d_in[17];
  const float* b2    = (const float*)d_in[18];
  const float* lnf_g = (const float*)d_in[19];
  const float* lnf_b = (const float*)d_in[20];
  const float* Wp    = (const float*)d_in[21];
  const float* bp    = (const float*)d_in[22];
  (void)in_sizes; (void)n_in; (void)out_size; (void)ws_size;

  char* base = (char*)d_ws;
  size_t off = 0;
  auto allocB = [&](size_t bytes) {
    void* p = base + off;
    off = (off + bytes + 255) & ~(size_t)255;
    return p;
  };
  float*  hbuf  = (float*)allocB((size_t)ROWS * D_MODEL * 4);          // residual
  float*  sbuf  = (float*)allocB((size_t)BATCH * N_HEADS * N_TOK * N_TOK * 4);
  __bf16* xbf   = (__bf16*)allocB((size_t)ROWS * VOCAB * 2);
  __bf16* wembT = (__bf16*)allocB((size_t)D_MODEL * VOCAB * 2);        // [384][8192]
  __bf16* wpT   = (__bf16*)allocB((size_t)VOCAB * D_MODEL * 2);        // [8192][384]
  __bf16* wqT   = (__bf16*)allocB((size_t)N_HEADS * HEAD * D_MODEL * 2);
  __bf16* wkT   = (__bf16*)allocB((size_t)N_HEADS * HEAD * D_MODEL * 2);
  __bf16* wvT   = (__bf16*)allocB((size_t)N_HEADS * HEAD * D_MODEL * 2);
  __bf16* woT   = (__bf16*)allocB((size_t)D_MODEL * D_MODEL * 2);
  __bf16* w1T   = (__bf16*)allocB((size_t)4 * D_MODEL * D_MODEL * 2);  // [1536][384]
  __bf16* w2T   = (__bf16*)allocB((size_t)D_MODEL * 4 * D_MODEL * 2);  // [384][1536]
  __bf16* ybf   = (__bf16*)allocB((size_t)ROWS * D_MODEL * 2);
  __bf16* qb    = (__bf16*)allocB((size_t)ROWS * D_MODEL * 2);         // [B,H,N,64]
  __bf16* kb    = (__bf16*)allocB((size_t)ROWS * D_MODEL * 2);         // [B,H,N,64]
  __bf16* vtb   = (__bf16*)allocB((size_t)ROWS * D_MODEL * 2);         // [B,H,64,N]
  __bf16* ob    = (__bf16*)allocB((size_t)ROWS * D_MODEL * 2);         // [B,N,384]
  __bf16* mb    = (__bf16*)allocB((size_t)ROWS * 4 * D_MODEL * 2);
  __bf16* pb    = (__bf16*)allocB((size_t)BATCH * N_HEADS * N_TOK * N_TOK * 2);

  const long long qkvCb = (long long)N_HEADS * N_TOK * HEAD;   // per-batch q/k/v
  const long long qkvCh = (long long)N_TOK * HEAD;             // per-head  q/k/v
  const long long scSb  = (long long)N_HEADS * N_TOK * N_TOK;
  const long long scSh  = (long long)N_TOK * N_TOK;

  // 0) one-time conversions: x -> bf16, Wemb/Wp -> transposed bf16
  convert_bf16_kernel<<<(ROWS * (VOCAB / 4) + 255) / 256, 256, 0, stream>>>(
      x, xbf, (long long)ROWS * VOCAB / 4);
  transpose_bf16_kernel<<<dim3(D_MODEL / 32, VOCAB / 32, 1), 256, 0, stream>>>(
      Wemb, wembT, VOCAB, D_MODEL, 0, 0);
  transpose_bf16_kernel<<<dim3(VOCAB / 32, D_MODEL / 32, 1), 256, 0, stream>>>(
      Wp, wpT, D_MODEL, VOCAB, 0, 0);

  // 1) h = x @ Wemb + bemb + posenc
  launch_gemm(stream, xbf, wembT, bemb, nullptr, hbuf,
              ROWS, D_MODEL, VOCAB, VOCAB, VOCAB, D_MODEL,
              1, 1, 0, 0, 0, 0, 0, 0, 0, 1.0f, EPI_POSENC, OUT_F32);

  for (int l = 0; l < N_LAYERS; ++l) {
    const long long wqkvSz = (long long)N_HEADS * D_MODEL * HEAD;
    // per-layer weight transposes (tiny)
    transpose_bf16_kernel<<<dim3(2, 12, N_HEADS), 256, 0, stream>>>(
        Wq + l * wqkvSz, wqT, D_MODEL, HEAD, (long long)D_MODEL * HEAD,
        (long long)HEAD * D_MODEL);
    transpose_bf16_kernel<<<dim3(2, 12, N_HEADS), 256, 0, stream>>>(
        Wk + l * wqkvSz, wkT, D_MODEL, HEAD, (long long)D_MODEL * HEAD,
        (long long)HEAD * D_MODEL);
    transpose_bf16_kernel<<<dim3(2, 12, N_HEADS), 256, 0, stream>>>(
        Wv + l * wqkvSz, wvT, D_MODEL, HEAD, (long long)D_MODEL * HEAD,
        (long long)HEAD * D_MODEL);
    transpose_bf16_kernel<<<dim3(12, 12, 1), 256, 0, stream>>>(
        Wo + (size_t)l * D_MODEL * D_MODEL, woT, D_MODEL, D_MODEL, 0, 0);
    transpose_bf16_kernel<<<dim3(48, 12, 1), 256, 0, stream>>>(
        W1 + (size_t)l * D_MODEL * 4 * D_MODEL, w1T, D_MODEL, 4 * D_MODEL, 0, 0);
    transpose_bf16_kernel<<<dim3(12, 48, 1), 256, 0, stream>>>(
        W2 + (size_t)l * 4 * D_MODEL * D_MODEL, w2T, 4 * D_MODEL, D_MODEL, 0, 0);

    // 2) y = LN1(h)  (bf16)
    layernorm_kernel<<<ROWS, 128, 0, stream>>>(hbuf, ln1_g + l * D_MODEL,
                                               ln1_b + l * D_MODEL, ybf, D_MODEL);
    // 3) q,k (bf16 [B,H,N,64]), v (bf16 transposed [B,H,64,N])
    launch_gemm(stream, ybf, wqT, bq + (size_t)l * N_HEADS * HEAD, nullptr, qb,
                N_TOK, HEAD, D_MODEL, D_MODEL, D_MODEL, HEAD,
                BATCH * N_HEADS, N_HEADS,
                (long long)N_TOK * D_MODEL, 0, 0, (long long)HEAD * D_MODEL,
                qkvCb, qkvCh, HEAD, 1.0f, EPI_BIAS, OUT_BF16);
    launch_gemm(stream, ybf, wkT, bk + (size_t)l * N_HEADS * HEAD, nullptr, kb,
                N_TOK, HEAD, D_MODEL, D_MODEL, D_MODEL, HEAD,
                BATCH * N_HEADS, N_HEADS,
                (long long)N_TOK * D_MODEL, 0, 0, (long long)HEAD * D_MODEL,
                qkvCb, qkvCh, HEAD, 1.0f, EPI_BIAS, OUT_BF16);
    launch_gemm(stream, ybf, wvT, bv + (size_t)l * N_HEADS * HEAD, nullptr, vtb,
                N_TOK, HEAD, D_MODEL, D_MODEL, D_MODEL, N_TOK,
                BATCH * N_HEADS, N_HEADS,
                (long long)N_TOK * D_MODEL, 0, 0, (long long)HEAD * D_MODEL,
                qkvCb, qkvCh, HEAD, 1.0f, EPI_BIAS, OUT_BF16T);
    // 4) scores = q @ k^T * 0.125  (k is already B^T form)
    launch_gemm(stream, qb, kb, nullptr, nullptr, sbuf,
                N_TOK, N_TOK, HEAD, HEAD, HEAD, N_TOK,
                BATCH * N_HEADS, N_HEADS,
                qkvCb, qkvCh, qkvCb, qkvCh, scSb, scSh, 0,
                0.125f, EPI_NONE, OUT_F32);
    // 5) softmax -> bf16 probs
    softmax_kernel<<<BATCH * N_HEADS * N_TOK, 256, 0, stream>>>(sbuf, pb, N_TOK);
    // 6) o = attn @ v, head-merged bf16 [B,N,384]
    launch_gemm(stream, pb, vtb, nullptr, nullptr, ob,
                N_TOK, HEAD, N_TOK, N_TOK, N_TOK, D_MODEL,
                BATCH * N_HEADS, N_HEADS,
                scSb, scSh, qkvCb, qkvCh,
                (long long)N_TOK * D_MODEL, (long long)HEAD, 0,
                1.0f, EPI_NONE, OUT_BF16);
    // 7) h = h + o @ Wo + bo
    launch_gemm(stream, ob, woT, bo + (size_t)l * D_MODEL, hbuf, hbuf,
                ROWS, D_MODEL, D_MODEL, D_MODEL, D_MODEL, D_MODEL,
                1, 1, 0, 0, 0, 0, 0, 0, 0, 1.0f, EPI_BIAS_RES, OUT_F32);
    // 8) y = LN2(h)
    layernorm_kernel<<<ROWS, 128, 0, stream>>>(hbuf, ln2_g + l * D_MODEL,
                                               ln2_b + l * D_MODEL, ybf, D_MODEL);
    // 9) m = gelu(y @ W1 + b1)  (bf16)
    launch_gemm(stream, ybf, w1T, b1 + (size_t)l * 4 * D_MODEL, nullptr, mb,
                ROWS, 4 * D_MODEL, D_MODEL, D_MODEL, D_MODEL, 4 * D_MODEL,
                1, 1, 0, 0, 0, 0, 0, 0, 0, 1.0f, EPI_BIAS_GELU, OUT_BF16);
    // 10) h = h + m @ W2 + b2
    launch_gemm(stream, mb, w2T, b2 + (size_t)l * D_MODEL, hbuf, hbuf,
                ROWS, D_MODEL, 4 * D_MODEL, 4 * D_MODEL, 4 * D_MODEL, D_MODEL,
                1, 1, 0, 0, 0, 0, 0, 0, 0, 1.0f, EPI_BIAS_RES, OUT_F32);
  }

  // 11) logits = LN_f(h) @ Wp + bp -> d_out (fp32)
  layernorm_kernel<<<ROWS, 128, 0, stream>>>(hbuf, lnf_g, lnf_b, ybf, D_MODEL);
  launch_gemm(stream, ybf, wpT, bp, nullptr, (float*)d_out,
              ROWS, VOCAB, D_MODEL, D_MODEL, D_MODEL, VOCAB,
              1, 1, 0, 0, 0, 0, 0, 0, 0, 1.0f, EPI_BIAS, OUT_F32);
}
